// Attention_77360950935992
// MI455X (gfx1250) — matrix operand order repaired and checked by reading, not compiled, not run on hardware
//
#include <hip/hip_runtime.h>
#include <hip/hip_bf16.h>

typedef _Float16 h16;
typedef __attribute__((ext_vector_type(8)))  _Float16 v8h;
typedef __attribute__((ext_vector_type(16))) _Float16 v16h;
typedef __attribute__((ext_vector_type(8)))  float    v8f;
typedef __attribute__((ext_vector_type(4)))  unsigned int u32x4;
typedef __attribute__((ext_vector_type(8)))  int i32x8;
typedef __attribute__((ext_vector_type(4)))  int i32x4;

#define B_  2
#define T_  2048
#define C_  1024
#define H_  16
#define HD_ 64

#define NEG_INF (-__builtin_inff())

__device__ __forceinline__ v8f wmma_f32_f16(v16h a, v16h b, v8f c) {
  // D = A(16x32 f16) * B(32x16 f16) + C(16x16 f32)
  return __builtin_amdgcn_wmma_f32_16x16x32_f16(false, a, false, b, (short)0, c,
                                                false, false);
}

// A fragment (16x32, f16) from row-major (M x K) storage.
// Lane L: row m = L&15; contiguous K run c0..c0+15 with c0 = (L>>4)*16.
__device__ __forceinline__ v16h load_a_frag(const h16* base, int ld, int lane) {
  const int m  = lane & 15;
  const int c0 = (lane >> 4) << 4;
  const h16* p = base + (size_t)m * ld + c0;
  v8h lo = *(const v8h*)(p);
  v8h hi = *(const v8h*)(p + 8);
  v16h f;
#pragma unroll
  for (int i = 0; i < 8; ++i) { f[i] = lo[i]; f[i + 8] = hi[i]; }
  return f;
}

// B fragment (32x16, f16) where B-transposed is stored row-major as (N x K):
// lane L: column n = L&15, contiguous K run k0 = (L>>4)*16 .. +15 -> 32B load.
__device__ __forceinline__ v16h load_b_fragT(const h16* base, int ld, int lane) {
  const int n  = lane & 15;
  const int k0 = (lane >> 4) << 4;
  return *(const v16h*)(base + (size_t)n * ld + k0);
}

// ---------------------------------------------------------------- TDM
// Issue one Tensor Data Mover load: 64 rows x 64 f16 tile, global row stride
// ldk (halves) -> dense LDS tile (row stride 64 halves). 2D descriptor.
__device__ __forceinline__ void tdm_load_tile_64x64(h16* lds_dst,
                                                    const h16* gsrc, int ldk) {
  unsigned long long ga = (unsigned long long)(size_t)gsrc;
  unsigned int lds_off  = (unsigned int)(size_t)lds_dst;  // addr[31:0] = LDS byte offset

  u32x4 g0;
  g0[0] = 1u;                                        // count=1, user descriptor
  g0[1] = lds_off;                                   // lds_addr (bytes)
  g0[2] = (unsigned int)(ga & 0xFFFFFFFFu);          // global_addr[31:0]
  g0[3] = ((unsigned int)(ga >> 32) & 0x01FFFFFFu)   // global_addr[56:32]
          | 0x80000000u;                             // type=2 ("image") in [127:126]

  i32x8 g1;
  g1[0] = 0x00010000;            // workgroup_mask=0, data_size=1 (2 bytes)
  g1[1] = 0;                     // no atomic barrier; tensor_dim0[15:0]=0
  g1[2] = 0x4000;                // tensor_dim0 = 1<<30 (hi16); tensor_dim1[15:0]=0
  g1[3] = (64 << 16) | 0x4000;   // tensor_dim1 = 1<<30 (hi16); tile_dim0 = 64
  g1[4] = 64;                    // tile_dim1 = 64, tile_dim2 = 0
  g1[5] = ldk;                   // tensor_dim0_stride[31:0] (halves)
  g1[6] = 0;                     // stride hi / dim1_stride lo
  g1[7] = 0;                     // dim1_stride hi
  i32x4 gz4 = {};
  i32x8 gz8 = {};
  __builtin_amdgcn_tensor_load_to_lds(g0, g1, gz4, gz4, gz8, 0);
}

// ---------------------------------------------------------------- converts
__global__ void k_f32_to_f16(const float* __restrict__ in, h16* __restrict__ out,
                             int n) {
  int i = blockIdx.x * blockDim.x + threadIdx.x;
  if (i < n) out[i] = (h16)in[i];
}

// ------------------------------------------------------------- NT GEMM
// D[M,N] (f32 rm) = A[M,K] (f16 rm) * W[N,K]^T (f16 rm)
// Block = 8 waves: 128(M) x 64(N) tile. The shared 64xK W strip is staged
// through LDS by the Tensor Data Mover (double buffered); A streams from
// global per wave (16 rows each). All fragments for a 64-deep K chunk are
// hoisted into registers, then the 8 WMMAs issue back-to-back.
__global__ void k_gemm_nt(const h16* __restrict__ A, const h16* __restrict__ W,
                          float* __restrict__ D, int M, int N, int K) {
  __shared__ h16 Bs[2][64 * 64];  // two 8KB 64(n) x 64(k) tiles

  const int lane = threadIdx.x & 31;
  const int wv   = threadIdx.x >> 5;
  const int tn   = N >> 6;
  const int bm   = blockIdx.x / tn;
  const int bn   = blockIdx.x % tn;

  const h16* a0 = A + ((size_t)bm * 128 + wv * 16) * K;
  const h16* w0 = W + (size_t)bn * 64 * K;
  const int nchunks = K >> 6;

  if (wv == 0) {
    tdm_load_tile_64x64(&Bs[0][0], w0, K);
    __builtin_amdgcn_s_wait_tensorcnt(0);
  }
  __syncthreads();

  v8f acc[4] = {};
  for (int c = 0; c < nchunks; ++c) {
    if (wv == 0 && c + 1 < nchunks)
      tdm_load_tile_64x64(&Bs[(c + 1) & 1][0], w0 + (c + 1) * 64, K);

    const h16* bs = &Bs[c & 1][0];

    // hoist all fragments for this chunk, then 8 back-to-back WMMAs
    v16h a0f = load_a_frag(a0 + c * 64,      K, lane);
    v16h a1f = load_a_frag(a0 + c * 64 + 32, K, lane);
    v16h bf[8];
#pragma unroll
    for (int nb = 0; nb < 4; ++nb) {
      bf[nb]     = load_b_fragT(bs + nb * 16 * 64,      64, lane);
      bf[nb + 4] = load_b_fragT(bs + nb * 16 * 64 + 32, 64, lane);
    }
#pragma unroll
    for (int nb = 0; nb < 4; ++nb) acc[nb] = wmma_f32_f16(a0f, bf[nb],     acc[nb]);
#pragma unroll
    for (int nb = 0; nb < 4; ++nb) acc[nb] = wmma_f32_f16(a1f, bf[nb + 4], acc[nb]);

    if (wv == 0 && c + 1 < nchunks)
      __builtin_amdgcn_s_wait_tensorcnt(0);  // next buffer landed
    __syncthreads();                         // publish next / retire current
  }

  const int hi = lane >> 4;
  const int n  = lane & 15;
  const int row0 = bm * 128 + wv * 16;
#pragma unroll
  for (int nb = 0; nb < 4; ++nb)
#pragma unroll
    for (int r = 0; r < 8; ++r)
      D[(size_t)(row0 + r + 8 * hi) * N + bn * 64 + nb * 16 + n] = acc[nb][r];
}

// ------------------------------------------------------- RoPE + repack
// mode 0: rope, scale (Q)  -> (B,H,T,HD) f16
// mode 1: rope, no scale(K)-> (B,H,T,HD) f16
// mode 2: plain convert (V)-> transposed (B,H,HD,T) f16
__global__ void k_rope_pack(const float* __restrict__ F,
                            const float* __restrict__ cosb,
                            const float* __restrict__ sinb,
                            h16* __restrict__ out, int mode, float scale) {
  int i = blockIdx.x * blockDim.x + threadIdx.x;  // one pair per thread
  const int dp = i & 31;
  const int h  = (i >> 5) & 15;
  const int t  = (i >> 9) & 2047;
  const int b  = i >> 20;
  if (b >= B_) return;

  const float* src = F + (size_t)(b * T_ + t) * C_ + h * HD_ + 2 * dp;
  float x0 = src[0], x1 = src[1];

  if (mode == 2) {
    h16* dst = out + ((size_t)((b * H_ + h) * HD_ + 2 * dp)) * T_ + t;
    dst[0]  = (h16)x0;
    dst[T_] = (h16)x1;
  } else {
    float c = cosb[t * 32 + dp], s = sinb[t * 32 + dp];
    float o0 = (x0 * c - x1 * s) * scale;
    float o1 = (x0 * s + x1 * c) * scale;
    h16* dst = out + ((size_t)((b * H_ + h) * T_ + t)) * HD_ + 2 * dp;
    dst[0] = (h16)o0;
    dst[1] = (h16)o1;
  }
}

// --------------------------------------------------- flash attention
// One wave per 16-query tile of one (b,h). Keys processed 32 at a time.
// Qh,Kh: (B,H,T,HD) f16 (Q pre-scaled by 1/8). Vt: (B,H,HD,T) f16.
// Oh out: (B,T,C) f16.
__global__ void k_flash(const h16* __restrict__ Qh, const h16* __restrict__ Kh,
                        const h16* __restrict__ Vt, h16* __restrict__ Oh) {
  __shared__ h16 ldsP[8][16 * 32];  // per-wave P tile (C-layout -> A-layout)

  const int lane = threadIdx.x & 31;
  const int wv   = threadIdx.x >> 5;
  const int gid  = blockIdx.x * 8 + wv;
  const int bh   = gid >> 7;   // 0..31
  const int qt   = gid & 127;  // query tile
  const int qm   = qt * 16;
  const int b    = bh >> 4;
  const int h    = bh & 15;

  const h16* Q = Qh + (size_t)bh * T_ * HD_;
  const h16* K = Kh + (size_t)bh * T_ * HD_;
  const h16* V = Vt + (size_t)bh * HD_ * T_;

  const v16h qa0 = load_a_frag(Q + (size_t)qm * HD_,      HD_, lane);
  const v16h qa1 = load_a_frag(Q + (size_t)qm * HD_ + 32, HD_, lane);

  v8f acc0 = {}, acc1 = {}, acc2 = {}, acc3 = {};
  float mrow[8], lrow[8];
#pragma unroll
  for (int r = 0; r < 8; ++r) { mrow[r] = NEG_INF; lrow[r] = 0.0f; }

  const int hi = lane >> 4;
  const int nn = lane & 15;
  const int nkb = (qm + 47) >> 5;  // 32-wide key blocks covering keys <= qm+15
  h16* P = ldsP[wv];

  for (int jb = 0; jb < nkb; ++jb) {
    const int kb = jb * 32;

    // S = Q K^T  (two 16x16 score tiles: keys kb..+15 and kb+16..+31)
    v16h kf0 = load_b_fragT(K + (size_t)kb * HD_,             HD_, lane);
    v16h kf1 = load_b_fragT(K + (size_t)kb * HD_ + 32,        HD_, lane);
    v16h kf2 = load_b_fragT(K + (size_t)(kb + 16) * HD_,      HD_, lane);
    v16h kf3 = load_b_fragT(K + (size_t)(kb + 16) * HD_ + 32, HD_, lane);
    v8f s0 = {}, s1 = {};
    s0 = wmma_f32_f16(qa0, kf0, s0);
    s0 = wmma_f32_f16(qa1, kf1, s0);
    s1 = wmma_f32_f16(qa0, kf2, s1);
    s1 = wmma_f32_f16(qa1, kf3, s1);

    if (kb + 31 > qm) {  // causal mask only touches the trailing block
#pragma unroll
      for (int r = 0; r < 8; ++r) {
        int row = qm + r + 8 * hi;
        if (kb + nn      > row) s0[r] = NEG_INF;
        if (kb + 16 + nn > row) s1[r] = NEG_INF;
      }
    }

    // online softmax: each row lives in 16 lanes of one accumulator slot
#pragma unroll
    for (int r = 0; r < 8; ++r) {
      float mx = fmaxf(s0[r], s1[r]);
      mx = fmaxf(mx, __shfl_xor(mx, 1));
      mx = fmaxf(mx, __shfl_xor(mx, 2));
      mx = fmaxf(mx, __shfl_xor(mx, 4));
      mx = fmaxf(mx, __shfl_xor(mx, 8));
      float newm  = fmaxf(mrow[r], mx);
      float alpha = __expf(mrow[r] - newm);
      mrow[r] = newm;
      float p0 = __expf(s0[r] - newm);
      float p1 = __expf(s1[r] - newm);
      float rs = p0 + p1;
      rs += __shfl_xor(rs, 1);
      rs += __shfl_xor(rs, 2);
      rs += __shfl_xor(rs, 4);
      rs += __shfl_xor(rs, 8);
      lrow[r] = lrow[r] * alpha + rs;
      acc0[r] *= alpha; acc1[r] *= alpha; acc2[r] *= alpha; acc3[r] *= alpha;
      int m = r + 8 * hi;
      P[m * 32 + nn]      = (h16)p0;
      P[m * 32 + nn + 16] = (h16)p1;
    }

    // O += P(16x32) * V(32x64); Vt row-major (HD,T) -> contiguous B frags
    v16h pa  = load_a_frag(P, 32, lane);
    v16h vf0 = load_b_fragT(V + (size_t)0  * T_ + kb, T_, lane);
    v16h vf1 = load_b_fragT(V + (size_t)16 * T_ + kb, T_, lane);
    v16h vf2 = load_b_fragT(V + (size_t)32 * T_ + kb, T_, lane);
    v16h vf3 = load_b_fragT(V + (size_t)48 * T_ + kb, T_, lane);
    acc0 = wmma_f32_f16(pa, vf0, acc0);
    acc1 = wmma_f32_f16(pa, vf1, acc1);
    acc2 = wmma_f32_f16(pa, vf2, acc2);
    acc3 = wmma_f32_f16(pa, vf3, acc3);
  }

  // normalize rows and emit f16 into (B,T,C) for the Wo projection
#pragma unroll
  for (int r = 0; r < 8; ++r) {
    float inv = 1.0f / lrow[r];
    int t = qm + r + 8 * hi;
    h16* dst = Oh + (size_t)(b * T_ + t) * C_ + h * HD_ + nn;
    dst[0]  = (h16)(acc0[r] * inv);
    dst[16] = (h16)(acc1[r] * inv);
    dst[32] = (h16)(acc2[r] * inv);
    dst[48] = (h16)(acc3[r] * inv);
  }
}

extern "C" void kernel_launch(void* const* d_in, const int* in_sizes, int n_in,
                              void* d_out, int out_size, void* d_ws, size_t ws_size,
                              hipStream_t stream) {
  (void)in_sizes; (void)n_in; (void)out_size; (void)ws_size;
  const float* x  = (const float*)d_in[0];
  const float* Wq = (const float*)d_in[1];
  const float* Wk = (const float*)d_in[2];
  const float* Wv = (const float*)d_in[3];
  const float* Wo = (const float*)d_in[4];
  const float* cs = (const float*)d_in[5];
  const float* sn = (const float*)d_in[6];
  float* out = (float*)d_out;

  char* ws = (char*)d_ws;
  const size_t MB = 1024u * 1024u;
  h16*   xh   = (h16*)(ws + 0);         // 8 MB: x as f16 (4096x1024)
  h16*   Wqh  = (h16*)(ws + 8 * MB);    // 2 MB each
  h16*   Wkh  = (h16*)(ws + 10 * MB);
  h16*   Wvh  = (h16*)(ws + 12 * MB);
  h16*   Woh  = (h16*)(ws + 14 * MB);
  float* Ftmp = (float*)(ws + 16 * MB); // 16 MB f32 GEMM scratch, reused 3x
  h16*   Qh   = (h16*)(ws + 32 * MB);   // 8 MB (B,H,T,HD)
  h16*   Kh   = (h16*)(ws + 40 * MB);   // 8 MB (B,H,T,HD)
  h16*   Vt   = (h16*)(ws + 48 * MB);   // 8 MB (B,H,HD,T)
  h16*   Oh   = (h16*)(ws + 56 * MB);   // 8 MB (B,T,C)

  const int M = B_ * T_, N = C_, K = C_;
  const int nconv_x = M * K, nconv_w = N * K;

  k_f32_to_f16<<<(nconv_x + 255) / 256, 256, 0, stream>>>(x,  xh,  nconv_x);
  k_f32_to_f16<<<(nconv_w + 255) / 256, 256, 0, stream>>>(Wq, Wqh, nconv_w);
  k_f32_to_f16<<<(nconv_w + 255) / 256, 256, 0, stream>>>(Wk, Wkh, nconv_w);
  k_f32_to_f16<<<(nconv_w + 255) / 256, 256, 0, stream>>>(Wv, Wvh, nconv_w);
  k_f32_to_f16<<<(nconv_w + 255) / 256, 256, 0, stream>>>(Wo, Woh, nconv_w);

  const int gemm_blocks = (M / 128) * (N / 64);  // 8 waves / block, 128x64 tile
  const int rope_blocks = (M * N / 2 + 255) / 256;

  // Q = rope(x Wq^T) * 1/sqrt(HD)
  k_gemm_nt<<<gemm_blocks, 256, 0, stream>>>(xh, Wqh, Ftmp, M, N, K);
  k_rope_pack<<<rope_blocks, 256, 0, stream>>>(Ftmp, cs, sn, Qh, 0, 0.125f);
  // K = rope(x Wk^T)
  k_gemm_nt<<<gemm_blocks, 256, 0, stream>>>(xh, Wkh, Ftmp, M, N, K);
  k_rope_pack<<<rope_blocks, 256, 0, stream>>>(Ftmp, cs, sn, Kh, 1, 1.0f);
  // V = x Wv^T, stored transposed
  k_gemm_nt<<<gemm_blocks, 256, 0, stream>>>(xh, Wvh, Ftmp, M, N, K);
  k_rope_pack<<<rope_blocks, 256, 0, stream>>>(Ftmp, cs, sn, Vt, 2, 1.0f);

  // causal flash attention
  k_flash<<<(B_ * H_ * (T_ / 16)) / 8, 256, 0, stream>>>(Qh, Kh, Vt, Oh);

  // out = O Wo^T (f32)
  k_gemm_nt<<<gemm_blocks, 256, 0, stream>>>(Oh, Woh, out, M, N, K);
}
